// GETModel_25237227831480
// MI455X (gfx1250) — matrix operand order repaired and checked by reading, not compiled, not run on hardware
//
#include <hip/hip_runtime.h>
#include <hip/hip_fp16.h>
#include <math.h>

typedef __attribute__((ext_vector_type(16))) _Float16 v16h;
typedef __attribute__((ext_vector_type(8)))  _Float16 v8h;
typedef __attribute__((ext_vector_type(8)))  float    v8f;

#define DD 128

// ---------------- device helpers ----------------

__device__ __forceinline__ float wave_reduce_sum(float v) {
#pragma unroll
  for (int off = 16; off > 0; off >>= 1) v += __shfl_xor(v, off, 32);
  return v;
}

__device__ __forceinline__ unsigned okey(float f) {
  unsigned u = __float_as_uint(f);
  return (u >> 31) ? ~u : (u | 0x80000000u);
}
__device__ __forceinline__ float odec(unsigned k) {
  unsigned u = (k >> 31) ? (k & 0x7FFFFFFFu) : ~k;
  return __uint_as_float(u);
}

__device__ __forceinline__ float softplusf(float x) {
  return x > 20.0f ? x : log1pf(__expf(x));
}

// ---------------- scalar coefficients ----------------
// cst[0]=b2/sqrt_d  cst[1]=-sp2/b2  cst[2]=-sp2/sqrt_d
// cst[3]=b3/D       cst[4]=-sp3/b3  cst[5]=-sp3/D
// cst[6]=bm/sqrt_d  cst[7]=-spm/bm  cst[8]=-spm/sqrt_d
__global__ void consts_kernel(const float* l2p, const float* l3p, const float* lmp,
                              const float* b2p, const float* b3p, const float* bmp,
                              float* cst, float* energy) {
  if (threadIdx.x == 0 && blockIdx.x == 0) {
    const float sqrt_d = 11.313708499f;
    float sp2 = softplusf(*l2p), sp3 = softplusf(*l3p), spm = softplusf(*lmp);
    float b2 = fminf(softplusf(*b2p), 5.0f);
    float b3 = fminf(softplusf(*b3p), 5.0f);
    float bm = fminf(softplusf(*bmp), 5.0f);
    cst[0] = b2 / sqrt_d;  cst[1] = -sp2 / b2;  cst[2] = -sp2 / sqrt_d;
    cst[3] = b3 / 128.0f;  cst[4] = -sp3 / b3;  cst[5] = -sp3 / 128.0f;
    cst[6] = bm / sqrt_d;  cst[7] = -spm / bm;  cst[8] = -spm / sqrt_d;
    *energy = 0.0f;
  }
}

// ---------------- layernorm forward (wave per row) ----------------
__global__ __launch_bounds__(256)
void ln_forward_kernel(const float* __restrict__ X, const float* __restrict__ gamma,
                       const float* __restrict__ beta, float* __restrict__ G,
                       _Float16* __restrict__ Gh, float* __restrict__ mu_o,
                       float* __restrict__ rstd_o, int N) {
  int row = blockIdx.x * (blockDim.x >> 5) + (threadIdx.x >> 5);
  int lane = threadIdx.x & 31;
  if (row >= N) return;
  float4 x4 = ((const float4*)(X + (size_t)row * DD))[lane];
  float mu = wave_reduce_sum(x4.x + x4.y + x4.z + x4.w) * (1.0f / 128.0f);
  float d0 = x4.x - mu, d1 = x4.y - mu, d2 = x4.z - mu, d3 = x4.w - mu;
  float var = wave_reduce_sum(d0 * d0 + d1 * d1 + d2 * d2 + d3 * d3) * (1.0f / 128.0f);
  float rstd = rsqrtf(var + 1e-5f);
  int c = lane * 4;
  float g0 = d0 * rstd * gamma[c + 0] + beta[c + 0];
  float g1 = d1 * rstd * gamma[c + 1] + beta[c + 1];
  float g2 = d2 * rstd * gamma[c + 2] + beta[c + 2];
  float g3 = d3 * rstd * gamma[c + 3] + beta[c + 3];
  ((float4*)(G + (size_t)row * DD))[lane] = make_float4(g0, g1, g2, g3);
  _Float16* gp = Gh + (size_t)row * DD + c;
  gp[0] = (_Float16)g0; gp[1] = (_Float16)g1; gp[2] = (_Float16)g2; gp[3] = (_Float16)g3;
  if (lane == 0) { mu_o[row] = mu; rstd_o[row] = rstd; }
}

// ---------------- f32 -> f16 convert ----------------
__global__ void f32_to_f16_kernel(const float* __restrict__ src,
                                  _Float16* __restrict__ dst, size_t n) {
  size_t i = (size_t)blockIdx.x * blockDim.x + threadIdx.x;
  if (i < n) dst[i] = (_Float16)src[i];
}

// ---------------- WMMA GEMM: C[M,Nc] (+)= A[M,K] @ B ----------------
// B operand staged in LDS as [K][Nc] row-major halves (transposed during staging
// for TRANSB). Each wave computes a 16(M) x 64(N) strip: 4 independent
// accumulators interleave WMMAs so no D->C hazard NOPs are needed, and the A
// fragment is reused 4x. K, Nc multiples of 32/64; K*Nc <= 16384.
template <int TRANSB, int ACCUM>
__global__ __launch_bounds__(256)
void wmma_gemm_kernel(const _Float16* __restrict__ A, const _Float16* __restrict__ B,
                      float* __restrict__ C, int M, int Nc, int K) {
  __shared__ __align__(32) _Float16 Bl[16384];
  int tid = threadIdx.x;
  int total = K * Nc;
  if (TRANSB) {
    // B stored [Nc,K]; want Bl[k*Nc+n] = B[n*K+k]; read contiguous, write strided
    for (int i = tid; i < total; i += blockDim.x) {
      int n = i / K, k = i - n * K;
      Bl[k * Nc + n] = B[i];
    }
  } else {
    for (int i = tid; i < total; i += blockDim.x) Bl[i] = B[i];
  }
  __syncthreads();

  int lane = tid & 31;
  int wid = blockIdx.x * (blockDim.x >> 5) + (tid >> 5);
  int NT4 = Nc >> 6;                 // groups of 4 n-tiles (64 cols)
  int MT = (M + 15) >> 4;
  if (wid >= MT * NT4) return;       // wave-uniform
  int mt = wid / NT4, ng = wid - mt * NT4;

  v8f acc0, acc1, acc2, acc3;
#pragma unroll
  for (int i = 0; i < 8; ++i) { acc0[i] = 0.0f; acc1[i] = 0.0f; acc2[i] = 0.0f; acc3[i] = 0.0f; }

  // Clamp instead of zero-masking: out-of-range A rows only affect C rows that
  // are masked at the store, so this is exact and avoids per-lane EXEC churn.
  int arow = mt * 16 + (lane & 15);
  if (arow >= M) arow = M - 1;
  const _Float16* arowp = A + (size_t)arow * K;
  int khalf = (lane >> 4) << 3;      // lanes 0-15 -> K base 0, lanes 16-31 -> +8

  for (int kb = 0; kb < K; kb += 32) {
    union { v16h v; v8h h[2]; } af;
    af.h[0] = *(const v8h*)(arowp + kb + khalf);
    af.h[1] = *(const v8h*)(arowp + kb + 16 + khalf);
    const _Float16* brow = Bl + (size_t)kb * Nc + ng * 64 + (lane & 15);  // lane%16 = N col
    v16h b0, b1, b2, b3;
#pragma unroll
    for (int s = 0; s < 16; ++s) {
      int kk = khalf + (s & 7) + ((s >> 3) << 4);
      const _Float16* bp = brow + (size_t)kk * Nc;
      b0[s] = bp[0]; b1[s] = bp[16]; b2[s] = bp[32]; b3[s] = bp[48];
    }
    acc0 = __builtin_amdgcn_wmma_f32_16x16x32_f16(false, af.v, false, b0, (short)0, acc0, false, false);
    acc1 = __builtin_amdgcn_wmma_f32_16x16x32_f16(false, af.v, false, b1, (short)0, acc1, false, false);
    acc2 = __builtin_amdgcn_wmma_f32_16x16x32_f16(false, af.v, false, b2, (short)0, acc2, false, false);
    acc3 = __builtin_amdgcn_wmma_f32_16x16x32_f16(false, af.v, false, b3, (short)0, acc3, false, false);
  }

  int crow0 = mt * 16 + ((lane >> 4) << 3);
  int ccol = ng * 64 + (lane & 15);
#pragma unroll
  for (int j = 0; j < 8; ++j) {
    int r = crow0 + j;
    if (r < M) {
      size_t base = (size_t)r * Nc + ccol;
      C[base +  0] = (ACCUM ? C[base +  0] : 0.0f) + acc0[j];
      C[base + 16] = (ACCUM ? C[base + 16] : 0.0f) + acc1[j];
      C[base + 32] = (ACCUM ? C[base + 32] : 0.0f) + acc2[j];
      C[base + 48] = (ACCUM ? C[base + 48] : 0.0f) + acc3[j];
    }
  }
}

// ---------------- edge MLP: a2[e] = gelu(ea@We1+be1)@We2 + be2 ----------------
__global__ __launch_bounds__(256)
void edge_mlp_kernel(const float* __restrict__ EA, const float* __restrict__ We1,
                     const float* __restrict__ be1, const float* __restrict__ We2,
                     const float* __restrict__ be2, float* __restrict__ a2, int E) {
  int e = blockIdx.x * blockDim.x + threadIdx.x;
  if (e >= E) return;
  float ea[16];
#pragma unroll
  for (int k = 0; k < 16; ++k) ea[k] = EA[(size_t)e * 16 + k];
  float acc = be2[0];
  for (int j = 0; j < 128; ++j) {
    float h = be1[j];
#pragma unroll
    for (int k = 0; k < 16; ++k) h += ea[k] * We1[k * 128 + j];
    float t = tanhf(0.7978845608f * (h + 0.044715f * h * h * h));
    acc += 0.5f * h * (1.0f + t) * We2[j];
  }
  a2[e] = acc;
}

// ---------------- order-2 scores (wave per edge) ----------------
__global__ __launch_bounds__(256)
void edge_score_kernel(const float* __restrict__ Q2, const float* __restrict__ K2,
                       const float* __restrict__ a2, const int* __restrict__ c2,
                       const int* __restrict__ u2, const float* __restrict__ cst,
                       float* __restrict__ s2, unsigned* __restrict__ mk, int E) {
  int e = blockIdx.x * (blockDim.x >> 5) + (threadIdx.x >> 5);
  int lane = threadIdx.x & 31;
  if (e >= E) return;
  int c = c2[e], u = u2[e];
  float4 q = ((const float4*)(Q2 + (size_t)c * DD))[lane];
  float4 k = ((const float4*)(K2 + (size_t)u * DD))[lane];
  float d = wave_reduce_sum(q.x * k.x + q.y * k.y + q.z * k.z + q.w * k.w);
  if (lane == 0) {
    float s = cst[0] * d + a2[e];
    s2[e] = s;
    atomicMax(&mk[c], okey(s));
  }
}

// ---------------- order-3 scores (wave per triplet) ----------------
__global__ __launch_bounds__(256)
void tri_score_kernel(const float* __restrict__ Q3, const float* __restrict__ K3,
                      const float* __restrict__ Tt, const int* __restrict__ c3,
                      const int* __restrict__ u3, const int* __restrict__ v3,
                      const int* __restrict__ ty, const float* __restrict__ cst,
                      float* __restrict__ s3, unsigned* __restrict__ mk, int T) {
  int t = blockIdx.x * (blockDim.x >> 5) + (threadIdx.x >> 5);
  int lane = threadIdx.x & 31;
  if (t >= T) return;
  int c = c3[t], u = u3[t], v = v3[t], tau = ty[t];
  float4 q = ((const float4*)(Q3 + (size_t)c * DD))[lane];
  float4 ku = ((const float4*)(K3 + (size_t)u * DD))[lane];
  float4 kv = ((const float4*)(K3 + (size_t)v * DD))[lane];
  float4 tv = ((const float4*)(Tt + (size_t)tau * DD))[lane];
  float d = wave_reduce_sum(q.x * ku.x * kv.x * tv.x + q.y * ku.y * kv.y * tv.y +
                            q.z * ku.z * kv.z * tv.z + q.w * ku.w * kv.w * tv.w);
  if (lane == 0) {
    float s = cst[3] * d;
    s3[t] = s;
    atomicMax(&mk[c], okey(s));
  }
}

// ---------------- segment sum of exp(s - max) ----------------
__global__ void seg_z_kernel(const float* __restrict__ s, const int* __restrict__ idx,
                             const unsigned* __restrict__ mk, float* __restrict__ z, int E) {
  int e = blockIdx.x * blockDim.x + threadIdx.x;
  if (e >= E) return;
  int c = idx[e];
  unsigned k = mk[c];
  float ms = (k == 0u) ? 0.0f : odec(k);
  atomicAdd(&z[c], __expf(s[e] - ms));
}

// ---------------- per-segment logsumexp + energy accumulation ----------------
__global__ void seg_lse_kernel(const unsigned* __restrict__ mk, const float* __restrict__ z,
                               float* __restrict__ lse, const float* __restrict__ cst,
                               int coefIdx, float* __restrict__ energy, int N) {
  int n = blockIdx.x * blockDim.x + threadIdx.x;
  if (n >= N) return;
  unsigned k = mk[n];
  float ms = (k == 0u) ? 0.0f : odec(k);
  float zz = z[n];
  float l = (zz > 0.0f) ? (logf(fmaxf(zz, 1e-30f)) + ms) : 0.0f;
  lse[n] = l;
  atomicAdd(energy, cst[coefIdx] * l);
}

// ---------------- memory-bank: scores + logsumexp + Pm (wave per row, Km in LDS) ----------------
__global__ __launch_bounds__(256)
void mem_kernel(const float* __restrict__ Qm, const float* __restrict__ Kmf,
                const float* __restrict__ cst, _Float16* __restrict__ Pmh,
                float* __restrict__ energy, int N) {
  __shared__ float sKm[32 * DD];
  for (int i = threadIdx.x; i < 32 * DD; i += blockDim.x) sKm[i] = Kmf[i];
  __syncthreads();
  int row = blockIdx.x * (blockDim.x >> 5) + (threadIdx.x >> 5);
  int lane = threadIdx.x & 31;
  if (row >= N) return;
  float4 q = ((const float4*)(Qm + (size_t)row * DD))[lane];
  float smscale = cst[6];
  float myscore = 0.0f, mx = -3.0e38f;
  for (int k = 0; k < 32; ++k) {
    float4 km = ((const float4*)(sKm + k * DD))[lane];
    float d = wave_reduce_sum(q.x * km.x + q.y * km.y + q.z * km.z + q.w * km.w);
    float s = smscale * d;  // butterfly reduce: identical on all lanes
    if (k == lane) myscore = s;
    mx = fmaxf(mx, s);
  }
  float zsum = wave_reduce_sum(__expf(myscore - mx));
  float lsem = mx + logf(zsum);
  if (lane == 0) atomicAdd(energy, cst[7] * lsem);
  Pmh[(size_t)row * 32 + lane] = (_Float16)(cst[8] * __expf(myscore - lsem));
}

// ---------------- order-2 backward scatter (wave per edge) ----------------
__global__ __launch_bounds__(256)
void edge_scatter_kernel(const float* __restrict__ Q2, const float* __restrict__ K2,
                         const float* __restrict__ s2, const float* __restrict__ lse2,
                         const int* __restrict__ c2, const int* __restrict__ u2,
                         const float* __restrict__ cst, float* __restrict__ dQ,
                         float* __restrict__ dK, int E) {
  int e = blockIdx.x * (blockDim.x >> 5) + (threadIdx.x >> 5);
  int lane = threadIdx.x & 31;
  if (e >= E) return;
  int c = c2[e], u = u2[e];
  float w = cst[2] * __expf(s2[e] - lse2[c]);
  float4 q = ((const float4*)(Q2 + (size_t)c * DD))[lane];
  float4 k = ((const float4*)(K2 + (size_t)u * DD))[lane];
  int col = lane * 4;
  float* dq = dQ + (size_t)c * DD + col;
  float* dk = dK + (size_t)u * DD + col;
  atomicAdd(dq + 0, w * k.x); atomicAdd(dq + 1, w * k.y);
  atomicAdd(dq + 2, w * k.z); atomicAdd(dq + 3, w * k.w);
  atomicAdd(dk + 0, w * q.x); atomicAdd(dk + 1, w * q.y);
  atomicAdd(dk + 2, w * q.z); atomicAdd(dk + 3, w * q.w);
}

// ---------------- order-3 backward scatter (wave per triplet) ----------------
__global__ __launch_bounds__(256)
void tri_scatter_kernel(const float* __restrict__ Q3, const float* __restrict__ K3,
                        const float* __restrict__ Tt, const float* __restrict__ s3,
                        const float* __restrict__ lse3, const int* __restrict__ c3,
                        const int* __restrict__ u3, const int* __restrict__ v3,
                        const int* __restrict__ ty, const float* __restrict__ cst,
                        float* __restrict__ dQ, float* __restrict__ dK, int T) {
  int t = blockIdx.x * (blockDim.x >> 5) + (threadIdx.x >> 5);
  int lane = threadIdx.x & 31;
  if (t >= T) return;
  int c = c3[t], u = u3[t], v = v3[t], tau = ty[t];
  float w = cst[5] * __expf(s3[t] - lse3[c]);
  float4 q = ((const float4*)(Q3 + (size_t)c * DD))[lane];
  float4 ku = ((const float4*)(K3 + (size_t)u * DD))[lane];
  float4 kv = ((const float4*)(K3 + (size_t)v * DD))[lane];
  float4 tv = ((const float4*)(Tt + (size_t)tau * DD))[lane];
  int col = lane * 4;
  float* dq = dQ + (size_t)c * DD + col;
  float* du = dK + (size_t)u * DD + col;
  float* dv = dK + (size_t)v * DD + col;
  atomicAdd(dq + 0, w * ku.x * kv.x * tv.x); atomicAdd(dq + 1, w * ku.y * kv.y * tv.y);
  atomicAdd(dq + 2, w * ku.z * kv.z * tv.z); atomicAdd(dq + 3, w * ku.w * kv.w * tv.w);
  atomicAdd(du + 0, w * q.x * kv.x * tv.x);  atomicAdd(du + 1, w * q.y * kv.y * tv.y);
  atomicAdd(du + 2, w * q.z * kv.z * tv.z);  atomicAdd(du + 3, w * q.w * kv.w * tv.w);
  atomicAdd(dv + 0, w * q.x * ku.x * tv.x);  atomicAdd(dv + 1, w * q.y * ku.y * tv.y);
  atomicAdd(dv + 2, w * q.z * ku.z * tv.z);  atomicAdd(dv + 3, w * q.w * ku.w * tv.w);
}

// ---------------- LN backward + clip + step + state clip (wave per row) ----------------
__global__ __launch_bounds__(256)
void ln_bwd_update_kernel(const float* __restrict__ X, const float* __restrict__ gamma,
                          const float* __restrict__ dG, const float* __restrict__ mu_a,
                          const float* __restrict__ rstd_a, float* __restrict__ out, int N) {
  int row = blockIdx.x * (blockDim.x >> 5) + (threadIdx.x >> 5);
  int lane = threadIdx.x & 31;
  if (row >= N) return;
  float4 x = ((const float4*)(X + (size_t)row * DD))[lane];
  float4 g = ((const float4*)(dG + (size_t)row * DD))[lane];
  float mu = mu_a[row], rs = rstd_a[row];
  int c = lane * 4;
  float xh0 = (x.x - mu) * rs, xh1 = (x.y - mu) * rs, xh2 = (x.z - mu) * rs, xh3 = (x.w - mu) * rs;
  float dy0 = g.x * gamma[c + 0], dy1 = g.y * gamma[c + 1];
  float dy2 = g.z * gamma[c + 2], dy3 = g.w * gamma[c + 3];
  float s1 = wave_reduce_sum(dy0 + dy1 + dy2 + dy3) * (1.0f / 128.0f);
  float s2 = wave_reduce_sum(dy0 * xh0 + dy1 * xh1 + dy2 * xh2 + dy3 * xh3) * (1.0f / 128.0f);
  float gx0 = rs * (dy0 - s1 - xh0 * s2), gx1 = rs * (dy1 - s1 - xh1 * s2);
  float gx2 = rs * (dy2 - s1 - xh2 * s2), gx3 = rs * (dy3 - s1 - xh3 * s2);
  float gn2 = wave_reduce_sum(gx0 * gx0 + gx1 * gx1 + gx2 * gx2 + gx3 * gx3);
  float gn = fmaxf(sqrtf(gn2), 1e-6f);
  float cs = fminf(1.0f / gn, 1.0f);
  float step = 0.1f * 0.9999f * cs;
  float xn0 = x.x - step * gx0, xn1 = x.y - step * gx1;
  float xn2 = x.z - step * gx2, xn3 = x.w - step * gx3;
  float sn2 = wave_reduce_sum(xn0 * xn0 + xn1 * xn1 + xn2 * xn2 + xn3 * xn3);
  float sn = fmaxf(sqrtf(sn2), 1e-6f);
  float c2s = fminf(10.0f / sn, 1.0f);
  ((float4*)(out + (size_t)row * DD))[lane] =
      make_float4(xn0 * c2s, xn1 * c2s, xn2 * c2s, xn3 * c2s);
}

// ---------------- host launch ----------------

extern "C" void kernel_launch(void* const* d_in, const int* in_sizes, int n_in,
                              void* d_out, int out_size, void* d_ws, size_t ws_size,
                              hipStream_t stream) {
  const float* X       = (const float*)d_in[0];
  const float* EA      = (const float*)d_in[1];
  const float* gamma   = (const float*)d_in[2];
  const float* beta    = (const float*)d_in[3];
  const float* W_Q2    = (const float*)d_in[4];
  const float* W_K2    = (const float*)d_in[5];
  const float* W_Q3    = (const float*)d_in[6];
  const float* W_K3    = (const float*)d_in[7];
  const float* T_tau   = (const float*)d_in[8];
  const float* W_Qm    = (const float*)d_in[9];
  const float* W_Km    = (const float*)d_in[10];
  const float* B_mem   = (const float*)d_in[11];
  const float* We1     = (const float*)d_in[12];
  const float* be1     = (const float*)d_in[13];
  const float* We2     = (const float*)d_in[14];
  const float* be2     = (const float*)d_in[15];
  const float* l2      = (const float*)d_in[16];
  const float* l3      = (const float*)d_in[17];
  const float* lm      = (const float*)d_in[18];
  const float* b2      = (const float*)d_in[19];
  const float* b3      = (const float*)d_in[20];
  const float* bm      = (const float*)d_in[21];
  const int*   c2i     = (const int*)d_in[22];
  const int*   u2i     = (const int*)d_in[23];
  const int*   c3i     = (const int*)d_in[24];
  const int*   u3i     = (const int*)d_in[25];
  const int*   v3i     = (const int*)d_in[26];
  const int*   tti     = (const int*)d_in[27];

  const int N  = in_sizes[0] / DD;
  const int E  = in_sizes[22];
  const int T  = in_sizes[24];
  const int KM = in_sizes[11] / DD;  // 32

  char* ws = (char*)d_ws;
  size_t off = 0;
  auto alloc = [&](size_t bytes) {
    size_t o = off;
    off = (off + bytes + 255) & ~(size_t)255;
    return o;
  };
  const size_t ND4 = (size_t)N * DD * 4, ND2 = (size_t)N * DD * 2;
  float*     cst   = (float*)(ws + alloc(64));
  float*     G     = (float*)(ws + alloc(ND4));
  _Float16*  Gh    = (_Float16*)(ws + alloc(ND2));
  float*     mu    = (float*)(ws + alloc((size_t)N * 4));
  float*     rstd  = (float*)(ws + alloc((size_t)N * 4));
  _Float16*  WQ2h  = (_Float16*)(ws + alloc(DD * DD * 2));
  _Float16*  WK2h  = (_Float16*)(ws + alloc(DD * DD * 2));
  _Float16*  WQ3h  = (_Float16*)(ws + alloc(DD * DD * 2));
  _Float16*  WK3h  = (_Float16*)(ws + alloc(DD * DD * 2));
  _Float16*  WQmh  = (_Float16*)(ws + alloc(DD * DD * 2));
  _Float16*  WKmh  = (_Float16*)(ws + alloc(DD * DD * 2));
  _Float16*  Bmh   = (_Float16*)(ws + alloc((size_t)KM * DD * 2));
  float*     Kmf   = (float*)(ws + alloc((size_t)KM * DD * 4));
  _Float16*  Kmh   = (_Float16*)(ws + alloc((size_t)KM * DD * 2));
  float*     Q2    = (float*)(ws + alloc(ND4));
  float*     K2    = (float*)(ws + alloc(ND4));
  float*     Q3    = (float*)(ws + alloc(ND4));
  float*     K3    = (float*)(ws + alloc(ND4));
  float*     Qm    = (float*)(ws + alloc(ND4));
  float*     a2    = (float*)(ws + alloc((size_t)E * 4));
  float*     s2    = (float*)(ws + alloc((size_t)E * 4));
  unsigned*  m2k   = (unsigned*)(ws + alloc((size_t)N * 4));
  float*     z2    = (float*)(ws + alloc((size_t)N * 4));
  float*     lse2  = (float*)(ws + alloc((size_t)N * 4));
  float*     s3    = (float*)(ws + alloc((size_t)T * 4));
  unsigned*  m3k   = (unsigned*)(ws + alloc((size_t)N * 4));
  float*     z3    = (float*)(ws + alloc((size_t)N * 4));
  float*     lse3  = (float*)(ws + alloc((size_t)N * 4));
  _Float16*  Pmh   = (_Float16*)(ws + alloc((size_t)N * KM * 2));
  float*     dG    = (float*)(ws + alloc(ND4));
  float*     dA    = (float*)(ws + alloc(ND4));
  float*     dB    = (float*)(ws + alloc(ND4));
  _Float16*  dAh   = (_Float16*)(ws + alloc(ND2));
  (void)ws_size; (void)n_in; (void)out_size;

  float* out    = (float*)d_out;
  float* energy = out + (size_t)N * DD;

  // zero accumulators
  hipMemsetAsync(m2k, 0, (size_t)N * 4, stream);
  hipMemsetAsync(z2,  0, (size_t)N * 4, stream);
  hipMemsetAsync(m3k, 0, (size_t)N * 4, stream);
  hipMemsetAsync(z3,  0, (size_t)N * 4, stream);
  hipMemsetAsync(dG,  0, ND4, stream);
  hipMemsetAsync(dA,  0, ND4, stream);
  hipMemsetAsync(dB,  0, ND4, stream);

  consts_kernel<<<1, 32, 0, stream>>>(l2, l3, lm, b2, b3, bm, cst, energy);
  ln_forward_kernel<<<(N + 7) / 8, 256, 0, stream>>>(X, gamma, beta, G, Gh, mu, rstd, N);

  const size_t WN = (size_t)DD * DD;
  f32_to_f16_kernel<<<(WN + 255) / 256, 256, 0, stream>>>(W_Q2, WQ2h, WN);
  f32_to_f16_kernel<<<(WN + 255) / 256, 256, 0, stream>>>(W_K2, WK2h, WN);
  f32_to_f16_kernel<<<(WN + 255) / 256, 256, 0, stream>>>(W_Q3, WQ3h, WN);
  f32_to_f16_kernel<<<(WN + 255) / 256, 256, 0, stream>>>(W_K3, WK3h, WN);
  f32_to_f16_kernel<<<(WN + 255) / 256, 256, 0, stream>>>(W_Qm, WQmh, WN);
  f32_to_f16_kernel<<<(WN + 255) / 256, 256, 0, stream>>>(W_Km, WKmh, WN);
  f32_to_f16_kernel<<<((size_t)KM * DD + 255) / 256, 256, 0, stream>>>(B_mem, Bmh, (size_t)KM * DD);

  auto gemm_blocks = [](int M, int Nc) {
    long strips = (long)((M + 15) / 16) * (Nc / 64);  // one 16x64 strip per wave
    return (int)((strips + 7) / 8);
  };
  // forward projections (NN, overwrite)
  wmma_gemm_kernel<0, 0><<<gemm_blocks(N, DD), 256, 0, stream>>>(Gh, WQ2h, Q2, N, DD, DD);
  wmma_gemm_kernel<0, 0><<<gemm_blocks(N, DD), 256, 0, stream>>>(Gh, WK2h, K2, N, DD, DD);
  wmma_gemm_kernel<0, 0><<<gemm_blocks(N, DD), 256, 0, stream>>>(Gh, WQ3h, Q3, N, DD, DD);
  wmma_gemm_kernel<0, 0><<<gemm_blocks(N, DD), 256, 0, stream>>>(Gh, WK3h, K3, N, DD, DD);
  wmma_gemm_kernel<0, 0><<<gemm_blocks(N, DD), 256, 0, stream>>>(Gh, WQmh, Qm, N, DD, DD);
  wmma_gemm_kernel<0, 0><<<gemm_blocks(KM, DD), 256, 0, stream>>>(Bmh, WKmh, Kmf, KM, DD, DD);
  f32_to_f16_kernel<<<((size_t)KM * DD + 255) / 256, 256, 0, stream>>>(Kmf, Kmh, (size_t)KM * DD);

  // order-2 energy
  edge_mlp_kernel<<<(E + 255) / 256, 256, 0, stream>>>(EA, We1, be1, We2, be2, a2, E);
  edge_score_kernel<<<(E + 7) / 8, 256, 0, stream>>>(Q2, K2, a2, c2i, u2i, cst, s2, m2k, E);
  seg_z_kernel<<<(E + 255) / 256, 256, 0, stream>>>(s2, c2i, m2k, z2, E);
  seg_lse_kernel<<<(N + 255) / 256, 256, 0, stream>>>(m2k, z2, lse2, cst, 1, energy, N);

  // order-3 energy
  tri_score_kernel<<<(T + 7) / 8, 256, 0, stream>>>(Q3, K3, T_tau, c3i, u3i, v3i, tti, cst, s3, m3k, T);
  seg_z_kernel<<<(T + 255) / 256, 256, 0, stream>>>(s3, c3i, m3k, z3, T);
  seg_lse_kernel<<<(N + 255) / 256, 256, 0, stream>>>(m3k, z3, lse3, cst, 4, energy, N);

  // memory-bank energy + Pm (scaled softmax) for backward
  mem_kernel<<<(N + 7) / 8, 256, 0, stream>>>(Qm, Kmf, cst, Pmh, energy, N);

  // backward: order-2 scatter then dG += dQ2@WQ2^T + dK2@WK2^T
  edge_scatter_kernel<<<(E + 7) / 8, 256, 0, stream>>>(Q2, K2, s2, lse2, c2i, u2i, cst, dA, dB, E);
  f32_to_f16_kernel<<<((size_t)N * DD + 255) / 256, 256, 0, stream>>>(dA, dAh, (size_t)N * DD);
  wmma_gemm_kernel<1, 1><<<gemm_blocks(N, DD), 256, 0, stream>>>(dAh, WQ2h, dG, N, DD, DD);
  f32_to_f16_kernel<<<((size_t)N * DD + 255) / 256, 256, 0, stream>>>(dB, dAh, (size_t)N * DD);
  wmma_gemm_kernel<1, 1><<<gemm_blocks(N, DD), 256, 0, stream>>>(dAh, WK2h, dG, N, DD, DD);

  // backward: order-3 scatter then dG += dQ3@WQ3^T + dK3@WK3^T
  hipMemsetAsync(dA, 0, ND4, stream);
  hipMemsetAsync(dB, 0, ND4, stream);
  tri_scatter_kernel<<<(T + 7) / 8, 256, 0, stream>>>(Q3, K3, T_tau, s3, lse3, c3i, u3i, v3i, tti,
                                                      cst, dA, dB, T);
  f32_to_f16_kernel<<<((size_t)N * DD + 255) / 256, 256, 0, stream>>>(dA, dAh, (size_t)N * DD);
  wmma_gemm_kernel<1, 1><<<gemm_blocks(N, DD), 256, 0, stream>>>(dAh, WQ3h, dG, N, DD, DD);
  f32_to_f16_kernel<<<((size_t)N * DD + 255) / 256, 256, 0, stream>>>(dB, dAh, (size_t)N * DD);
  wmma_gemm_kernel<1, 1><<<gemm_blocks(N, DD), 256, 0, stream>>>(dAh, WK3h, dG, N, DD, DD);

  // backward: memory term — dQm = Pm_scaled @ Km (NN, K=32), then dG += dQm@WQm^T
  wmma_gemm_kernel<0, 0><<<gemm_blocks(N, DD), 256, 0, stream>>>(Pmh, Kmh, dA, N, DD, KM);
  f32_to_f16_kernel<<<((size_t)N * DD + 255) / 256, 256, 0, stream>>>(dA, dAh, (size_t)N * DD);
  wmma_gemm_kernel<1, 1><<<gemm_blocks(N, DD), 256, 0, stream>>>(dAh, WQmh, dG, N, DD, DD);

  // LN backward + clip + step + state clip
  ln_bwd_update_kernel<<<(N + 7) / 8, 256, 0, stream>>>(X, gamma, dG, mu, rstd, out, N);
}
